// EdgeAwareMPBlock_60138132078613
// MI455X (gfx1250) — matrix operand order repaired and checked by reading, not compiled, not run on hardware
//
#include <hip/hip_runtime.h>

#define N 512
#define D 128
#define PITCH_A 136   // bf16 elements per A-tile row in LDS (272B rows -> 4-bank skew)

typedef __attribute__((ext_vector_type(16))) __bf16 v16bf;
typedef __attribute__((ext_vector_type(2)))  __bf16 bf16x2;
typedef __attribute__((ext_vector_type(2)))  float  f32x2;
typedef __attribute__((ext_vector_type(8)))  float  v8f;
typedef unsigned int u32x4 __attribute__((ext_vector_type(4)));
typedef int          i32x8 __attribute__((ext_vector_type(8)));
typedef int          i32x4 __attribute__((ext_vector_type(4)));

union FragAB { int4 q[2]; v16bf v; };

#if defined(__HIP_DEVICE_COMPILE__) && __has_builtin(__builtin_amdgcn_tensor_load_to_lds)
#define HAVE_TDM 1
#else
#define HAVE_TDM 0
#endif

// One v_cvt_pk_bf16_f32 per dword, no repack fixups.
__device__ __forceinline__ unsigned int pack2bf(float a, float b) {
  f32x2 f = {a, b};
  bf16x2 h = __builtin_convertvector(f, bf16x2);
  return __builtin_bit_cast(unsigned int, h);
}

__device__ __forceinline__ float silu_f(float x) {
  // x * v_rcp_f32(1 + exp(-x)) : avoids the IEEE v_div_scale expansion
#if defined(__HIP_DEVICE_COMPILE__) && __has_builtin(__builtin_amdgcn_rcpf)
  return x * __builtin_amdgcn_rcpf(1.0f + __expf(-x));
#else
  return x / (1.0f + __expf(-x));
#endif
}

#if HAVE_TDM
// 1-D TDM copy: nf32 contiguous f32 from global -> LDS (ISA §8.3/8.4 D# layout).
__device__ __forceinline__ void tdm_load_1d(void* lds_dst, const void* gsrc, unsigned nf32) {
  unsigned long long ga = (unsigned long long)gsrc;
  unsigned ldso = (unsigned)(unsigned long long)lds_dst;   // addr[31:0] == LDS offset
  u32x4 g0;
  g0[0] = 1u;                                        // count=1, user descriptor
  g0[1] = ldso;                                      // lds_addr
  g0[2] = (unsigned)ga;                              // global_addr[31:0]
  g0[3] = (unsigned)((ga >> 32) & 0x01ffffffu)       // global_addr[56:32]
        | (2u << 30);                                // type = 2 ("image")
  i32x8 g1;
  g1[0] = 0x20000;                                   // data_size = 4B
  g1[1] = (int)((nf32 & 0xffffu) << 16);             // tensor_dim0 lo16
  g1[2] = (int)(((nf32 >> 16) & 0xffffu) | (1u << 16)); // tensor_dim0 hi16 | tensor_dim1=1
  g1[3] = (int)((nf32 & 0xffffu) << 16);             // tile_dim0
  g1[4] = 1;                                         // tile_dim1 = 1
  g1[5] = (int)nf32;                                 // tensor_dim0_stride lo32
  g1[6] = 0;
  g1[7] = 0;
  i32x4 z4 = {0, 0, 0, 0};
#if defined(__clang_major__) && (__clang_major__ >= 23)
  __builtin_amdgcn_tensor_load_to_lds(g0, g1, z4, z4, (i32x8)(0), 0);
#else
  __builtin_amdgcn_tensor_load_to_lds(g0, g1, z4, z4, 0);
#endif
}
#endif

__device__ __forceinline__ void store_a_tile(unsigned short* aTb, int row, int c4, float4 v) {
  unsigned int* dst = (unsigned int*)&aTb[row * PITCH_A + c4 * 4];
  dst[0] = pack2bf(v.x, v.y);
  dst[1] = pack2bf(v.z, v.w);
}

// ---------------------------------------------------------------------------
// Kernel 0a: pre-swizzle w1_e (= msg_w1 rows 128..255) into the per-lane
// B-fragment order used here for V_WMMA_F32_16X16X32_BF16 (contiguous k; lane = column, lane half =
// K offset +16, dword r = bf16 pair K=2r,2r+1).
// ---------------------------------------------------------------------------
__global__ __launch_bounds__(256) void prep_bswz(const float* __restrict__ msg_w1,
                                                 unsigned int* __restrict__ bswz) {
  int idx = blockIdx.x * blockDim.x + threadIdx.x;
  if (idx >= 8192) return;
  int r    = idx & 7;
  int lane = (idx >> 3) & 31;
  int kb   = (idx >> 8) & 3;
  int t    = idx >> 10;
  int k = kb * 32 + ((lane >> 4) * 16) + 2 * r;
  int c = t * 16 + (lane & 15);
  float a = msg_w1[(size_t)(D + k) * D + c];
  float b = msg_w1[(size_t)(D + k + 1) * D + c];
  bswz[idx] = pack2bf(a, b);
}

// ---------------------------------------------------------------------------
// Kernel 0b: hjb[j][d] = h[j] @ w1_h + msg_b1[d]
// ---------------------------------------------------------------------------
__global__ __launch_bounds__(128) void prep_hjb(const float* __restrict__ h,
                                                const float* __restrict__ msg_w1,
                                                const float* __restrict__ msg_b1,
                                                float* __restrict__ hjb) {
  __shared__ float sh[D];
  int j = blockIdx.x, t = threadIdx.x;
  sh[t] = h[(size_t)j * D + t];
  __syncthreads();
  float acc = msg_b1[t];
#pragma unroll 4
  for (int k = 0; k < D; ++k) acc += sh[k] * msg_w1[(size_t)k * D + t];
  hjb[(size_t)j * D + t] = acc;
}

// ---------------------------------------------------------------------------
// Kernel 1 (heavy): s[i] = sum_j mask[j,i] * silu(rel_emb[j,i]@w1_e + hjb[j])
// One block per i; 8 waves; wave w owns output columns 16w..16w+15.
// Double-buffered LDS + register-staged global loads; TDM for the hjb tile.
// ---------------------------------------------------------------------------
__global__ __launch_bounds__(256) void msg_main(const float* __restrict__ rel_emb,
                                                const int* __restrict__ edge_mask,
                                                const float* __restrict__ hjb,
                                                const unsigned int* __restrict__ bswz,
                                                float* __restrict__ sOut) {
  __shared__ unsigned short aT[2][16 * PITCH_A];  // bf16 A tiles
  __shared__ float hT[2][16 * D];                 // f32 hjb tiles (TDM target)
  __shared__ float mT[2][16];                     // mask weights

  const int i    = blockIdx.x;
  const int tid  = threadIdx.x;
  const int wave = tid >> 5;
  const int lane = tid & 31;
  const int lg   = lane >> 4;
  const int ln15 = lane & 15;

  // Per-thread staging coords: thread covers (row0,c4) and (row0+8,c4) float4s.
  const int row0 = tid >> 5;
  const int row1 = row0 + 8;
  const int c4   = tid & 31;

#define REL_PTR(j0, row) (rel_emb + (((size_t)((j0) + (row))) * N + i) * D + c4 * 4)

  // B fragments resident in registers for the whole kernel.
  FragAB bfr[4];
#pragma unroll
  for (int kb = 0; kb < 4; ++kb) {
    const int4* bp = (const int4*)(bswz + (size_t)wave * 1024 + kb * 256 + lane * 8);
    bfr[kb].q[0] = bp[0];
    bfr[kb].q[1] = bp[1];
  }

  // ---- prologue: stage chunk 0 into buffer 0 ----
  {
    float4 a0 = *(const float4*)REL_PTR(0, row0);
    float4 a1 = *(const float4*)REL_PTR(0, row1);
#if HAVE_TDM
    if (wave == 0) tdm_load_1d(&hT[0][0], hjb, 16 * D);
#else
    float4 h0 = *(const float4*)(hjb + (size_t)row0 * D + c4 * 4);
    float4 h1 = *(const float4*)(hjb + (size_t)row1 * D + c4 * 4);
#endif
    int m = (tid < 16) ? edge_mask[(size_t)tid * N + i] : 0;
    store_a_tile(&aT[0][0], row0, c4, a0);
    store_a_tile(&aT[0][0], row1, c4, a1);
#if !HAVE_TDM
    *(float4*)&hT[0][row0 * D + c4 * 4] = h0;
    *(float4*)&hT[0][row1 * D + c4 * 4] = h1;
#endif
    if (tid < 16) mT[0][tid] = (float)m;
#if HAVE_TDM
#if __has_builtin(__builtin_amdgcn_s_wait_tensorcnt)
    if (wave == 0) __builtin_amdgcn_s_wait_tensorcnt(0);
#endif
#endif
    __syncthreads();
  }

  float accS = 0.0f;

  for (int c = 0; c < 32; ++c) {
    const int buf  = c & 1;
    const int nbuf = buf ^ 1;
    const int j0n  = (c + 1) * 16;
    const bool more = (c + 1 < 32);

    // ---- issue next chunk's loads early (overlap with WMMA below) ----
    float4 na0, na1;
#if !HAVE_TDM
    float4 nh0, nh1;
#endif
    int nm = 0;
    if (more) {
      na0 = *(const float4*)REL_PTR(j0n, row0);
      na1 = *(const float4*)REL_PTR(j0n, row1);
#if HAVE_TDM
      if (wave == 0) tdm_load_1d(&hT[nbuf][0], hjb + (size_t)j0n * D, 16 * D);
#else
      nh0 = *(const float4*)(hjb + (size_t)(j0n + row0) * D + c4 * 4);
      nh1 = *(const float4*)(hjb + (size_t)(j0n + row1) * D + c4 * 4);
#endif
      if (tid < 16) nm = edge_mask[((size_t)(j0n + tid)) * N + i];
      if (c + 2 < 32) {
        __builtin_prefetch(REL_PTR((c + 2) * 16, row0), 0, 1);
        __builtin_prefetch(REL_PTR((c + 2) * 16, row1), 0, 1);
      }
    }

    // ---- WMMA on current buffer: pre = A(16x128) x B(128x16), f32 acc ----
    v8f cacc = {0.f, 0.f, 0.f, 0.f, 0.f, 0.f, 0.f, 0.f};
#pragma unroll
    for (int kb = 0; kb < 4; ++kb) {
      FragAB af;
      const int off = ln15 * PITCH_A + kb * 32 + lg * 16;  // 16B-aligned
      af.q[0] = *(const int4*)&aT[buf][off];
      af.q[1] = *(const int4*)&aT[buf][off + 8];
      cacc = __builtin_amdgcn_wmma_f32_16x16x32_bf16(false, af.v, false, bfr[kb].v,
                                                     (short)0, cacc, false, false);
    }

    // ---- + hjb, SiLU, mask, accumulate over the 16 j-rows ----
#pragma unroll
    for (int v = 0; v < 8; ++v) {
      int row = v + 8 * lg;   // C layout: VGPR v -> M=v (lo half) / M=v+8 (hi half)
      float x = cacc[v] + hT[buf][row * D + wave * 16 + ln15];
      accS += mT[buf][row] * silu_f(x);
    }

    __syncthreads();   // everyone done reading buffers for chunk c

    if (more) {
      store_a_tile(&aT[nbuf][0], row0, c4, na0);
      store_a_tile(&aT[nbuf][0], row1, c4, na1);
#if !HAVE_TDM
      *(float4*)&hT[nbuf][row0 * D + c4 * 4] = nh0;
      *(float4*)&hT[nbuf][row1 * D + c4 * 4] = nh1;
#endif
      if (tid < 16) mT[nbuf][tid] = (float)nm;
#if HAVE_TDM
#if __has_builtin(__builtin_amdgcn_s_wait_tensorcnt)
      if (wave == 0) __builtin_amdgcn_s_wait_tensorcnt(0);
#endif
#endif
    }
    __syncthreads();   // next buffer fully staged
  }
#undef REL_PTR

  // Combine the two lane halves (same column, disjoint row sets).
  accS += __shfl_xor(accS, 16, 32);
  if (lane < 16) sOut[(size_t)i * D + wave * 16 + lane] = accS;
}

// ---------------------------------------------------------------------------
// Kernel 2: agg = s@W2 + cnt*b2 ; update MLP ; residual + LayerNorm.
// ---------------------------------------------------------------------------
__global__ __launch_bounds__(128) void update_kernel(const float* __restrict__ h,
                                                     const int* __restrict__ edge_mask,
                                                     const float* __restrict__ sIn,
                                                     const float* __restrict__ msg_w2,
                                                     const float* __restrict__ msg_b2,
                                                     const float* __restrict__ upd_w1,
                                                     const float* __restrict__ upd_b1,
                                                     const float* __restrict__ upd_w2,
                                                     const float* __restrict__ upd_b2,
                                                     const float* __restrict__ ln_g,
                                                     const float* __restrict__ ln_b,
                                                     float* __restrict__ out) {
  __shared__ float shS[D], shH[D], shA[D], shT[D], red[D];
  const int i = blockIdx.x;
  const int t = threadIdx.x;

  shS[t] = sIn[(size_t)i * D + t];
  shH[t] = h[(size_t)i * D + t];
  float pc = 0.0f;
#pragma unroll
  for (int p = 0; p < 4; ++p) pc += (float)edge_mask[((size_t)(t + 128 * p)) * N + i];
  red[t] = pc;
  __syncthreads();
  for (int s = 64; s > 0; s >>= 1) { if (t < s) red[t] += red[t + s]; __syncthreads(); }
  const float cnt = red[0];
  __syncthreads();

  // agg[t] = s . W2[:,t] + cnt * b2[t]
  float acc = cnt * msg_b2[t];
#pragma unroll 4
  for (int k = 0; k < D; ++k) acc += shS[k] * msg_w2[(size_t)k * D + t];
  shA[t] = acc;
  __syncthreads();

  // hidden = silu([h, agg] @ upd_w1 + b1)
  float u = upd_b1[t];
#pragma unroll 4
  for (int k = 0; k < D; ++k) u += shH[k] * upd_w1[(size_t)k * D + t];
#pragma unroll 4
  for (int k = 0; k < D; ++k) u += shA[k] * upd_w1[(size_t)(D + k) * D + t];
  shT[t] = silu_f(u);
  __syncthreads();

  // delta = hidden @ upd_w2 + b2 ; x = h + delta
  float delta = upd_b2[t];
#pragma unroll 4
  for (int k = 0; k < D; ++k) delta += shT[k] * upd_w2[(size_t)k * D + t];
  const float x = shH[t] + delta;

  // LayerNorm (biased variance, eps=1e-5)
  red[t] = x;
  __syncthreads();
  for (int s = 64; s > 0; s >>= 1) { if (t < s) red[t] += red[t + s]; __syncthreads(); }
  const float mu = red[0] * (1.0f / 128.0f);
  __syncthreads();
  const float dxm = x - mu;
  red[t] = dxm * dxm;
  __syncthreads();
  for (int s = 64; s > 0; s >>= 1) { if (t < s) red[t] += red[t + s]; __syncthreads(); }
  const float var = red[0] * (1.0f / 128.0f);
  out[(size_t)i * D + t] = dxm * rsqrtf(var + 1e-5f) * ln_g[t] + ln_b[t];
}

// ---------------------------------------------------------------------------
extern "C" void kernel_launch(void* const* d_in, const int* in_sizes, int n_in,
                              void* d_out, int out_size, void* d_ws, size_t ws_size,
                              hipStream_t stream) {
  const float* h        = (const float*)d_in[0];
  const float* rel_emb  = (const float*)d_in[1];
  const int*   edge_msk = (const int*)d_in[2];
  const float* msg_w1   = (const float*)d_in[3];
  const float* msg_b1   = (const float*)d_in[4];
  const float* msg_w2   = (const float*)d_in[5];
  const float* msg_b2   = (const float*)d_in[6];
  const float* upd_w1   = (const float*)d_in[7];
  const float* upd_b1   = (const float*)d_in[8];
  const float* upd_w2   = (const float*)d_in[9];
  const float* upd_b2   = (const float*)d_in[10];
  const float* ln_g     = (const float*)d_in[11];
  const float* ln_b     = (const float*)d_in[12];
  float* out = (float*)d_out;

  // Workspace layout: [B_swz: 32KB][hjb: 256KB][s: 256KB]
  unsigned int* bswz = (unsigned int*)d_ws;
  float* hjb = (float*)((char*)d_ws + 32768);
  float* sWS = (float*)((char*)d_ws + 32768 + (size_t)N * D * 4);

  prep_bswz<<<32, 256, 0, stream>>>(msg_w1, bswz);
  prep_hjb<<<N, 128, 0, stream>>>(h, msg_w1, msg_b1, hjb);
  msg_main<<<N, 256, 0, stream>>>(rel_emb, edge_msk, hjb, bswz, sWS);
  update_kernel<<<N, 128, 0, stream>>>(h, edge_msk, sWS, msg_w2, msg_b2,
                                       upd_w1, upd_b1, upd_w2, upd_b2,
                                       ln_g, ln_b, out);
}
